// SpatioTemporalXORRouter_1468878815291
// MI455X (gfx1250) — 8-bit matrix operand order repaired and checked by reading, compiled, not run on hardware
//
#include <hip/hip_runtime.h>
#include <hip/hip_bf16.h>
#include <stdint.h>

typedef __attribute__((ext_vector_type(8))) int v8i;

#define SIG_DIM    512
#define NUM_ATOMS  64
#define KCHUNKS    8
#define NTILES     4
#define FRAG_DWORDS 256
#define NFRAGS     64
#define WS_FRAG_DWORDS (NFRAGS * FRAG_DWORDS)
#define WS_COLSUM_OFF  WS_FRAG_DWORDS

__device__ __forceinline__ float bspline(float x) {
    float t = fabsf(x);
    float near_ = 2.0f / 3.0f - t * t + 0.5f * t * t * t;
    float c = 2.0f - t;
    float far_ = (c * c * c) / 6.0f;
    return (t < 1.0f) ? near_ : ((t < 2.0f) ? far_ : 0.0f);
}

__global__ __launch_bounds__(1024) void pack_sigs_kernel(
    const float* __restrict__ sigs, uint32_t* __restrict__ ws)
{
    int tid  = threadIdx.x;
    int lane = tid & 31;
    int f    = tid >> 5;
    int kchunk = f >> 2;
    int ntile  = f & 3;
    int a    = ntile * 16 + (lane & 15);
    int half = lane >> 4;

    #pragma unroll
    for (int j = 0; j < 8; ++j) {
        uint32_t pd = 0, nd = 0;
        int kbase = kchunk * 64 + (j >> 2) * 32 + half * 16 + (j & 3) * 4;
        #pragma unroll
        for (int t = 0; t < 4; ++t) {
            float s = sigs[a * SIG_DIM + kbase + t];
            pd |= (s > 0.0f ? 1u : 0u) << (8 * t);
            nd |= (s < 0.0f ? 1u : 0u) << (8 * t);
        }
        ws[(size_t)f        * FRAG_DWORDS + lane * 8 + j] = pd;
        ws[(size_t)(f + 32) * FRAG_DWORDS + lane * 8 + j] = nd;
    }

    if (tid < NUM_ATOMS) {
        int c = 0;
        for (int k = 0; k < SIG_DIM; ++k)
            c += (sigs[tid * SIG_DIM + k] != 0.0f) ? 1 : 0;
        ((int*)ws)[WS_COLSUM_OFF + tid] = c;
    }
}

__device__ __forceinline__ void packA(const float* __restrict__ crow,
                                      int kc, int half,
                                      v8i& ap, v8i& an, int& nzcnt)
{
    #pragma unroll
    for (int j = 0; j < 8; ++j) {
        int K0 = kc * 64 + (j >> 2) * 32 + half * 16 + (j & 3) * 4;
        float4 x = *(const float4*)(crow + K0);
        float xs[4] = {x.x, x.y, x.z, x.w};
        uint32_t pd = 0, nd = 0;
        #pragma unroll
        for (int t = 0; t < 4; ++t) {
            pd |= (xs[t] > 0.0f ? 1u : 0u) << (8 * t);
            nd |= (xs[t] < 0.0f ? 1u : 0u) << (8 * t);
        }
        ap[j] = (int)pd;
        an[j] = (int)nd;
        nzcnt += (int)(((pd | nd) * 0x01010101u) >> 24);
    }
}

__global__ __launch_bounds__(256) void router_kernel(
    const float* __restrict__ content,
    const int*   __restrict__ position,
    const int*   __restrict__ state,
    const float* __restrict__ atom_pos,
    const float* __restrict__ comp,
    const uint32_t* __restrict__ ws,
    float* __restrict__ out,
    int Brows, int ntiles_total)
{
    __shared__ uint4 smem[WS_FRAG_DWORDS / 4];

    int tid = threadIdx.x;
    const uint4* ws4 = (const uint4*)ws;
    #pragma unroll
    for (int i = 0; i < (WS_FRAG_DWORDS / 4) / 256; ++i)
        smem[tid + i * 256] = ws4[tid + i * 256];
    __syncthreads();

    int wave = tid >> 5;
    int lane = tid & 31;
    int tile = blockIdx.x * 8 + wave;
    if (tile >= ntiles_total) return;

    int mrow = lane & 15;
    int half = lane >> 4;

    v8i accp[NTILES], accn[NTILES];
    #pragma unroll
    for (int nt = 0; nt < NTILES; ++nt) {
        accp[nt] = v8i{0, 0, 0, 0, 0, 0, 0, 0};
        accn[nt] = v8i{0, 0, 0, 0, 0, 0, 0, 0};
    }

    int nzcnt = 0;
    const float* crow = content + (size_t)(tile * 16 + mrow) * SIG_DIM;

    v8i apb[2], anb[2];
    packA(crow, 0, half, apb[0], anb[0], nzcnt);

    #pragma unroll
    for (int kc = 0; kc < KCHUNKS; ++kc) {
        int cur = kc & 1;
        if (kc + 1 < KCHUNKS)
            packA(crow, kc + 1, half, apb[cur ^ 1], anb[cur ^ 1], nzcnt);

        #pragma unroll
        for (int nt = 0; nt < NTILES; ++nt) {
            int fp = kc * 4 + nt;
            const uint4* bp4 = &smem[fp * 64 + lane * 2];
            uint4 l0 = bp4[0], h0 = bp4[1];
            v8i bp = {(int)l0.x, (int)l0.y, (int)l0.z, (int)l0.w,
                      (int)h0.x, (int)h0.y, (int)h0.z, (int)h0.w};
            const uint4* bn4 = &smem[(fp + 32) * 64 + lane * 2];
            uint4 l1 = bn4[0], h1 = bn4[1];
            v8i bn = {(int)l1.x, (int)l1.y, (int)l1.z, (int)l1.w,
                      (int)h1.x, (int)h1.y, (int)h1.z, (int)h1.w};
            accp[nt] = __builtin_amdgcn_wmma_i32_16x16x64_iu8(
                false, apb[cur], false, bp, accp[nt], false, false);
            accn[nt] = __builtin_amdgcn_wmma_i32_16x16x64_iu8(
                false, anb[cur], false, bn, accn[nt], false, false);
        }
    }

    int n0 = lane & 15;
    const int* colsum = (const int*)ws + WS_COLSUM_OFF;
    int   cs_a[NTILES];
    float ap_a[NTILES];
    #pragma unroll
    for (int nt = 0; nt < NTILES; ++nt) {
        int a = n0 + nt * 16;
        cs_a[nt] = colsum[a];
        ap_a[nt] = atom_pos[a];
    }

    #pragma unroll
    for (int v = 0; v < 8; ++v) {
        int rloc = v + half * 8;
        int r = tile * 16 + rloc;
        int rc = __shfl(nzcnt, rloc, 32) + __shfl(nzcnt, rloc + 16, 32);
        float pos = (float)position[r];

        float best  = -3.0e38f;
        int   besta = NUM_ATOMS;
        #pragma unroll
        for (int nt = 0; nt < NTILES; ++nt) {
            int a = n0 + nt * 16;
            int dot = accp[nt][v] + accn[nt][v];
            float cscore = (float)(2 * dot - cs_a[nt] - rc);
            float sp = bspline((pos - ap_a[nt]) * 0.5f);
            float sc = cscore + 10.0f * sp;
            if (sc > best || (sc == best && a < besta)) { best = sc; besta = a; }
        }
        #pragma unroll
        for (int m = 1; m <= 8; m <<= 1) {
            float ob = __shfl_xor(best, m, 32);
            int   oa = __shfl_xor(besta, m, 32);
            if (ob > best || (ob == best && oa < besta)) { best = ob; besta = oa; }
        }
        if (n0 == 0) {
            out[r] = (float)besta;
            int st = state[r];
            out[Brows + r] = (float)(int)comp[st * NUM_ATOMS + besta];
        }
    }
}

extern "C" void kernel_launch(void* const* d_in, const int* in_sizes, int n_in,
                              void* d_out, int out_size, void* d_ws, size_t ws_size,
                              hipStream_t stream)
{
    const float* content  = (const float*)d_in[0];
    const int*   position = (const int*)  d_in[1];
    const int*   state    = (const int*)  d_in[2];
    const float* sigs     = (const float*)d_in[3];
    const float* apos     = (const float*)d_in[4];
    const float* comp     = (const float*)d_in[5];

    int B = in_sizes[1];
    uint32_t* ws = (uint32_t*)d_ws;

    pack_sigs_kernel<<<1, 1024, 0, stream>>>(sigs, ws);

    int ntiles = B / 16;
    int blocks = (ntiles + 7) / 8;
    router_kernel<<<blocks, 256, 0, stream>>>(
        content, position, state, apos, comp, ws,
        (float*)d_out, B, ntiles);
}
